// AnatomicalGCN_30511447671393
// MI455X (gfx1250) — matrix operand order repaired and checked by reading, compiled, not run on hardware
//
#include <hip/hip_runtime.h>
#include <math.h>

typedef unsigned short u16;
typedef unsigned int   u32;
typedef __attribute__((ext_vector_type(16))) __bf16 v16bf;
typedef __attribute__((ext_vector_type(8)))  float  v8f;

#define B_   16
#define T_   256
#define BT   4096
#define NREG 6
#define RF   64
#define FDIM 128
#define NH   4
#define FF   2048
#define NTOK (BT * NREG)   // 24576

// ---------------------------------------------------------------------------
// helpers
// ---------------------------------------------------------------------------
__device__ __forceinline__ u16 f2bf(float x) {
  u32 u = __float_as_uint(x);
  u32 r = (u + 0x7FFFu + ((u >> 16) & 1u)) >> 16;
  return (u16)r;
}
__device__ __forceinline__ int imin(int a, int b) { return a < b ? a : b; }
__device__ __forceinline__ int imax(int a, int b) { return a > b ? a : b; }

#define WMMA_BF16(a, b, c) \
  __builtin_amdgcn_wmma_f32_16x16x32_bf16(false, (a), false, (b), (short)0, (c), false, false)

// A fragment: 16x32 bf16 tile, rows row0..+15, cols k0..+31, row-major LDS.
// Per documented layout each lane holds two contiguous 8-element runs ->
// two ds_load_b128.  Requires stride % 8 == 0 and k0 % 8 == 0.
__device__ __forceinline__ v16bf lds_frag_a(const u16* p, int row0, int k0, int stride, int lane) {
  int r = lane & 15, h = lane >> 4;
  const u16* base = p + (row0 + r) * stride + k0 + h * 8;
  union { v16bf v; uint4 q[2]; } f;
  f.q[0] = *(const uint4*)(base);
  f.q[1] = *(const uint4*)(base + 16);
  return f.v;
}

// B fragment: 32x16 bf16 tile read from a TRANSPOSED LDS tile Bt[n][k]
// (n-major, stride % 8 == 0).  Lane L (h = L>>4) needs Bt[n0+(L&15)][k0+8h..+7]
// and Bt[n0+(L&15)][k0+16+8h..+7] -> two ds_load_b128 (same k order as A).
__device__ __forceinline__ v16bf lds_frag_bT(const u16* pT, int k0, int n0, int stride, int lane) {
  int r = lane & 15, h = lane >> 4;
  const u16* base = pT + (n0 + r) * stride + k0 + h * 8;
  union { v16bf v; uint4 q[2]; } f;
  f.q[0] = *(const uint4*)(base);
  f.q[1] = *(const uint4*)(base + 16);
  return f.v;
}

// ---------------------------------------------------------------------------
// f32 -> bf16 convert
// ---------------------------------------------------------------------------
__global__ void k_f32_to_bf16(const float* __restrict__ in, u16* __restrict__ out, int n) {
  int i = blockIdx.x * blockDim.x + threadIdx.x;
  if (i < n) out[i] = f2bf(in[i]);
}

// ---------------------------------------------------------------------------
// Generic WMMA GEMM: C[M,N] = act(A[M,K] @ B[K,N] + bias)
// M%64==0, N%64==0, K%32==0. A,B bf16 row-major, C f32.
// ---------------------------------------------------------------------------
__launch_bounds__(256)
__global__ void k_gemm_bf16(const u16* __restrict__ A, const u16* __restrict__ Bm,
                            const float* __restrict__ bias, float* __restrict__ C,
                            int M, int N, int K, int relu_flag) {
  __shared__ __align__(16) u16 As[64 * 40];   // row-major, stride 40
  __shared__ __align__(16) u16 Bs[64 * 40];   // TRANSPOSED: Bs[n][k], stride 40
  int tid = threadIdx.x, lane = tid & 31, w = tid >> 5;
  int row0 = blockIdx.y * 64, col0 = blockIdx.x * 64;
  int m0 = (w & 3) * 16;
  int n0a = ((w >> 2) * 2) * 16, n0b = n0a + 16;
  v8f acc0 = {}, acc1 = {};

  for (int k0 = 0; k0 < K; k0 += 32) {
    {
      int r = tid >> 2, c = (tid & 3) * 8;
      uint4 v = *(const uint4*)(A + (size_t)(row0 + r) * K + k0 + c);
      *(uint4*)&As[r * 40 + c] = v;           // 40*2=80B row stride -> 16B aligned
    }
    {
      int r = tid >> 3, c = (tid & 7) * 8;    // global row k0+r, cols col0+c..c+7
      union { uint4 q; u16 s[8]; } v;
      v.q = *(const uint4*)(Bm + (size_t)(k0 + r) * N + col0 + c);
#pragma unroll
      for (int i = 0; i < 8; ++i) Bs[(c + i) * 40 + r] = v.s[i];   // transpose
    }
    __syncthreads();
    v16bf a  = lds_frag_a(As, m0, 0, 40, lane);
    v16bf b0 = lds_frag_bT(Bs, 0, n0a, 40, lane);
    v16bf b1 = lds_frag_bT(Bs, 0, n0b, 40, lane);
    acc0 = WMMA_BF16(a, b0, acc0);
    acc1 = WMMA_BF16(a, b1, acc1);
    __syncthreads();
  }
  int n = lane & 15, h = lane >> 4;
#pragma unroll
  for (int v = 0; v < 8; ++v) {
    int m = v + 8 * h;
    float e0 = acc0[v] + (bias ? bias[col0 + n0a + n] : 0.f);
    float e1 = acc1[v] + (bias ? bias[col0 + n0b + n] : 0.f);
    if (relu_flag) { e0 = fmaxf(e0, 0.f); e1 = fmaxf(e1, 0.f); }
    C[(size_t)(row0 + m0 + m) * N + col0 + n0a + n] = e0;
    C[(size_t)(row0 + m0 + m) * N + col0 + n0b + n] = e1;
  }
}

// ---------------------------------------------------------------------------
// Fused FFN + residual + LayerNorm (post-norm transformer FFN block):
//   Out = LN( Xres + relu(X@W1 + b1) @ W2 + b2 )
// X bf16 [M,D], W1 bf16 [D,FF], W2 bf16 [FF,D].  D in {64,128}, M%64==0.
// The [64 x 2048] hidden tile never leaves LDS.
// ---------------------------------------------------------------------------
template <int D>
__launch_bounds__(256)
__global__ void k_ffn_fused(const u16* __restrict__ X, const float* __restrict__ Xres,
                            const u16* __restrict__ W1, const float* __restrict__ B1,
                            const u16* __restrict__ W2, const float* __restrict__ B2,
                            const float* __restrict__ lnG, const float* __restrict__ lnB,
                            float* __restrict__ Out) {
  constexpr int XS  = D + 8;        // X tile stride (row-major), mult of 8
  constexpr int W1S = D + 8;        // W1 chunk TRANSPOSED: [64 n][D k]
  constexpr int W2S = 72;           // W2 chunk TRANSPOSED: [D n][64 k]
  constexpr int HSS = 72;           // hidden tile (64 x 64) row-major
  constexpr int WB  = (64 * W1S > D * W2S) ? 64 * W1S : D * W2S;
  constexpr int SMEM = 64 * XS * 2 + WB * 2 + 64 * HSS * 2;
  static_assert(SMEM >= 64 * D * 4, "LDS reuse for f32 out tile");
  __shared__ __align__(16) char smem[SMEM];
  u16* Xs = (u16*)smem;
  u16* Wb = Xs + 64 * XS;           // shared by W1t / W2t (phase-separated)
  u16* Hs = Wb + WB;
  float* Os = (float*)smem;         // reused after the main loop

  int tid = threadIdx.x, lane = tid & 31, w = tid >> 5;
  int row0 = blockIdx.x * 64;
  constexpr int NT = D / 32;        // output n-tiles per wave
  v8f zero = {};
  v8f oacc[NT];
#pragma unroll
  for (int t = 0; t < NT; ++t) oacc[t] = zero;

  // stage X tile (64 x D), row-major
  for (int i = tid; i < 64 * D / 8; i += 256) {
    int r = i / (D / 8), c = (i % (D / 8)) * 8;
    uint4 v = *(const uint4*)(X + (size_t)(row0 + r) * D + c);
    *(uint4*)&Xs[r * XS + c] = v;   // XS*2 = 144/272 bytes -> 16B aligned
  }

  int m0 = (w & 3) * 16;
  int n0a = ((w >> 2) * 2) * 16, n0b = n0a + 16;

  for (int cc = 0; cc < FF; cc += 64) {
    __syncthreads();
    // load W1[:, cc:cc+64] (D x 64) transposed into Wb: Wb[n][k], stride W1S
    for (int i = tid; i < D * 8; i += 256) {
      int r = i >> 3, c = (i & 7) * 8;          // global row r (k), cols cc+c..
      union { uint4 q; u16 s[8]; } v;
      v.q = *(const uint4*)(W1 + (size_t)r * FF + cc + c);
#pragma unroll
      for (int j = 0; j < 8; ++j) Wb[(c + j) * W1S + r] = v.s[j];
    }
    if (cc + 64 < FF) {
      __builtin_prefetch(W1 + (size_t)(tid >> 3) * FF + cc + 64, 0, 1);
      __builtin_prefetch(W2 + (size_t)(cc + 64 + (tid >> 3)) * D, 0, 1);
    }
    __syncthreads();
    // H = relu(X @ W1c + b1c)   (64 x 64)
    v8f h0 = zero, h1 = zero;
#pragma unroll
    for (int ks = 0; ks < D; ks += 32) {
      v16bf a  = lds_frag_a(Xs, m0, ks, XS, lane);
      v16bf b0 = lds_frag_bT(Wb, ks, n0a, W1S, lane);
      v16bf b1 = lds_frag_bT(Wb, ks, n0b, W1S, lane);
      h0 = WMMA_BF16(a, b0, h0);
      h1 = WMMA_BF16(a, b1, h1);
    }
    __syncthreads();
    {
      int n = lane & 15, hh = lane >> 4;
#pragma unroll
      for (int v = 0; v < 8; ++v) {
        int m = v + 8 * hh;
        Hs[(m0 + m) * HSS + n0a + n] = f2bf(fmaxf(h0[v] + B1[cc + n0a + n], 0.f));
        Hs[(m0 + m) * HSS + n0b + n] = f2bf(fmaxf(h1[v] + B1[cc + n0b + n], 0.f));
      }
    }
    // load W2[cc:cc+64, :] (64 x D) transposed into Wb: Wb[n][k], stride W2S
    for (int i = tid; i < 64 * D / 8; i += 256) {
      int r = i / (D / 8), c = (i % (D / 8)) * 8;  // global row cc+r (k), cols c..
      union { uint4 q; u16 s[8]; } v;
      v.q = *(const uint4*)(W2 + (size_t)(cc + r) * D + c);
#pragma unroll
      for (int j = 0; j < 8; ++j) Wb[(c + j) * W2S + r] = v.s[j];
    }
    __syncthreads();
    // OUT += H @ W2c
    v16bf a0 = lds_frag_a(Hs, m0, 0, HSS, lane);
    v16bf a1 = lds_frag_a(Hs, m0, 32, HSS, lane);
#pragma unroll
    for (int t = 0; t < NT; ++t) {
      int n0 = ((w >> 2) * NT + t) * 16;
      v16bf b0 = lds_frag_bT(Wb, 0, n0, W2S, lane);
      oacc[t] = WMMA_BF16(a0, b0, oacc[t]);
      v16bf b1 = lds_frag_bT(Wb, 32, n0, W2S, lane);
      oacc[t] = WMMA_BF16(a1, b1, oacc[t]);
    }
  }
  __syncthreads();
  // spill output tile (+bias) to LDS as f32
  {
    int n = lane & 15, hh = lane >> 4;
#pragma unroll
    for (int t = 0; t < NT; ++t) {
      int n0 = ((w >> 2) * NT + t) * 16;
#pragma unroll
      for (int v = 0; v < 8; ++v) {
        int m = v + 8 * hh;
        Os[(m0 + m) * D + n0 + n] = oacc[t][v] + B2[n0 + n];
      }
    }
  }
  __syncthreads();
  // residual + LayerNorm, one row per thread
  if (tid < 64) {
    const float* res = Xres + (size_t)(row0 + tid) * D;
    float s = 0.f, ss = 0.f;
    for (int d = 0; d < D; ++d) { float v = Os[tid * D + d] + res[d]; s += v; ss += v * v; }
    float mu = s / D, inv = rsqrtf(ss / D - mu * mu + 1e-5f);
    float* o = Out + (size_t)(row0 + tid) * D;
    for (int d = 0; d < D; ++d) {
      float v = Os[tid * D + d] + res[d];
      o[d] = (v - mu) * inv * lnG[d] + lnB[d];
    }
  }
}

// ---------------------------------------------------------------------------
// Per-region 2-layer GCN + mean/max pooling + LayerNorm -> 64-dim tokens.
// One wave32 per (frame, region); adjacency recomputed analytically.
// ---------------------------------------------------------------------------
__launch_bounds__(256)
__global__ void k_gcn(const float* __restrict__ x0, const float* __restrict__ x1,
                      const float* __restrict__ x2, const float* __restrict__ x3,
                      const float* __restrict__ x4, const float* __restrict__ x5,
                      const float* __restrict__ W1, const float* __restrict__ Bb1,
                      const float* __restrict__ W2, const float* __restrict__ Bb2,
                      const float* __restrict__ lnG, const float* __restrict__ lnB,
                      float* __restrict__ tokF, u16* __restrict__ tokH) {
  __shared__ float xbuf[8][220];
  __shared__ float h1buf[8][20 * 32];
  __shared__ float red[8][66];
  int tid = threadIdx.x, lu = tid >> 5, lane = tid & 31;
  int u = blockIdx.x * 8 + lu;                 // token row (frame*6 + region)
  int fr = u / 6, r = u % 6;
  const int ns[6] = {9, 9, 11, 11, 9, 20};
  const float* xs[6] = {x0, x1, x2, x3, x4, x5};
  int n = ns[r];
  const float* xr = xs[r] + (size_t)fr * n * 11;
  for (int i = lane; i < n * 11; i += 32) xbuf[lu][i] = xr[i];
  __syncthreads();

  int f = lane;                                 // feature 0..31
  const float* w1 = W1 + r * 11 * 32;
  float rs[20], t1[20];
  for (int i = 0; i < n; ++i)
    rs[i] = rsqrtf((float)(1 + imin(i, 2) + imin(n - 1 - i, 2)));
  for (int m = 0; m < n; ++m) {
    float s = 0.f;
#pragma unroll
    for (int c = 0; c < 11; ++c) s += xbuf[lu][m * 11 + c] * w1[c * 32 + f];
    t1[m] = s;
  }
  float b1v = Bb1[r * 32 + f];
  for (int i = 0; i < n; ++i) {
    float s = 0.f;
    int j0 = imax(0, i - 2), j1 = imin(n - 1, i + 2);
    for (int j = j0; j <= j1; ++j) s += rs[i] * rs[j] * t1[j];
    h1buf[lu][i * 32 + f] = fmaxf(s + b1v, 0.f);
  }
  __syncthreads();

  const float* w2 = W2 + r * 32 * 32;
  float b2v = Bb2[r * 32 + f];
  for (int m = 0; m < n; ++m) {
    float s = 0.f;
#pragma unroll
    for (int c = 0; c < 32; ++c) s += h1buf[lu][m * 32 + c] * w2[c * 32 + f];
    t1[m] = s;
  }
  float mn = 0.f, mx = -3.0e38f;
  for (int i = 0; i < n; ++i) {
    float s = 0.f;
    int j0 = imax(0, i - 2), j1 = imin(n - 1, i + 2);
    for (int j = j0; j <= j1; ++j) s += rs[i] * rs[j] * t1[j];
    float h = fmaxf(s + b2v, 0.f);
    mn += h; mx = fmaxf(mx, h);
  }
  mn /= (float)n;
  red[lu][f] = mn; red[lu][32 + f] = mx;
  __syncthreads();
  if (lane == 0) {
    float s = 0.f, ss = 0.f;
    for (int k = 0; k < 64; ++k) { float v = red[lu][k]; s += v; ss += v * v; }
    float mu = s / 64.f;
    red[lu][64] = mu;
    red[lu][65] = rsqrtf(ss / 64.f - mu * mu + 1e-5f);
  }
  __syncthreads();
  float mu = red[lu][64], inv = red[lu][65];
  size_t base = (size_t)u * 64;
  float v0 = (red[lu][f] - mu) * inv * lnG[r * 64 + f] + lnB[r * 64 + f];
  float v1 = (red[lu][32 + f] - mu) * inv * lnG[r * 64 + 32 + f] + lnB[r * 64 + 32 + f];
  tokF[base + f] = v0;       tokF[base + 32 + f] = v1;
  tokH[base + f] = f2bf(v0); tokH[base + 32 + f] = f2bf(v1);
}

// ---------------------------------------------------------------------------
// Spatial self-attention over 6 tokens per frame (thread = frame,head,query)
// qkv layout [token][Q(64)|K(64)|V(64)]
// ---------------------------------------------------------------------------
__global__ void k_spatial_attn(const float* __restrict__ qkv, u16* __restrict__ outH) {
  int idx = blockIdx.x * blockDim.x + threadIdx.x;
  if (idx >= BT * NH * 6) return;
  int qi = idx % 6, h = (idx / 6) % NH, fr = idx / (6 * NH);
  const float* q = qkv + (size_t)(fr * 6 + qi) * 192 + h * 16;
  float sc[6], mx = -3.0e38f;
  for (int j = 0; j < 6; ++j) {
    const float* k = qkv + (size_t)(fr * 6 + j) * 192 + 64 + h * 16;
    float s = 0.f;
#pragma unroll
    for (int d = 0; d < 16; ++d) s += q[d] * k[d];
    sc[j] = s * 0.25f;
    mx = fmaxf(mx, sc[j]);
  }
  float den = 0.f;
  for (int j = 0; j < 6; ++j) { sc[j] = __expf(sc[j] - mx); den += sc[j]; }
  float o[16];
#pragma unroll
  for (int d = 0; d < 16; ++d) o[d] = 0.f;
  for (int j = 0; j < 6; ++j) {
    float wv = sc[j] / den;
    const float* v = qkv + (size_t)(fr * 6 + j) * 192 + 128 + h * 16;
#pragma unroll
    for (int d = 0; d < 16; ++d) o[d] += wv * v[d];
  }
  u16* op = outH + (size_t)(fr * 6 + qi) * 64 + h * 16;
#pragma unroll
  for (int d = 0; d < 16; ++d) op[d] = f2bf(o[d]);
}

// ---------------------------------------------------------------------------
// Cross-attention: 1 mean query attends over the 6 region tokens
// ---------------------------------------------------------------------------
__global__ void k_ca_attn(const float* __restrict__ qq, const float* __restrict__ kv,
                          u16* __restrict__ outH) {
  int idx = blockIdx.x * blockDim.x + threadIdx.x;
  if (idx >= BT * NH) return;
  int h = idx % NH, fr = idx / NH;
  const float* q = qq + (size_t)fr * 192 + h * 16;
  float sc[6], mx = -3.0e38f;
  for (int j = 0; j < 6; ++j) {
    const float* k = kv + (size_t)(fr * 6 + j) * 192 + 64 + h * 16;
    float s = 0.f;
#pragma unroll
    for (int d = 0; d < 16; ++d) s += q[d] * k[d];
    sc[j] = s * 0.25f;
    mx = fmaxf(mx, sc[j]);
  }
  float den = 0.f;
  for (int j = 0; j < 6; ++j) { sc[j] = __expf(sc[j] - mx); den += sc[j]; }
  float o[16];
#pragma unroll
  for (int d = 0; d < 16; ++d) o[d] = 0.f;
  for (int j = 0; j < 6; ++j) {
    float wv = sc[j] / den;
    const float* v = kv + (size_t)(fr * 6 + j) * 192 + 128 + h * 16;
#pragma unroll
    for (int d = 0; d < 16; ++d) o[d] += wv * v[d];
  }
  u16* op = outH + (size_t)fr * 64 + h * 16;
#pragma unroll
  for (int d = 0; d < 16; ++d) op[d] = f2bf(o[d]);
}

// ---------------------------------------------------------------------------
// Temporal attention, online softmax, one thread per query (block = (b,h))
// qkv layout [row][Q(128)|K(128)|V(128)], head dim 32
// ---------------------------------------------------------------------------
__launch_bounds__(256)
__global__ void k_temporal_attn(const float* __restrict__ qkv, u16* __restrict__ outH) {
  int bh = blockIdx.x;            // 0..63
  int b = bh >> 2, h = bh & 3;
  int t = threadIdx.x;
  const float* q = qkv + (size_t)(b * T_ + t) * 384 + h * 32;
  float qr[32];
#pragma unroll
  for (int d = 0; d < 32; ++d) qr[d] = q[d];
  float m = -3.0e38f, l = 0.f, acc[32];
#pragma unroll
  for (int d = 0; d < 32; ++d) acc[d] = 0.f;
  const float scale = 0.17677669529663687f;     // 1/sqrt(32)
  for (int j = 0; j < T_; ++j) {
    const float* kp = qkv + (size_t)(b * T_ + j) * 384 + 128 + h * 32;
    float s = 0.f;
#pragma unroll
    for (int d = 0; d < 32; ++d) s += qr[d] * kp[d];
    s *= scale;
    float mn = fmaxf(m, s);
    float corr = __expf(m - mn);
    float p = __expf(s - mn);
    l = l * corr + p;
    const float* vp = qkv + (size_t)(b * T_ + j) * 384 + 256 + h * 32;
#pragma unroll
    for (int d = 0; d < 32; ++d) acc[d] = acc[d] * corr + p * vp[d];
    m = mn;
  }
  float invl = 1.f / l;
  u16* op = outH + (size_t)(b * T_ + t) * 128 + h * 32;
#pragma unroll
  for (int d = 0; d < 32; ++d) op[d] = f2bf(acc[d] * invl);
}

// ---------------------------------------------------------------------------
// out = LN(x + res); optional bf16 copy
// ---------------------------------------------------------------------------
__global__ void k_ln_residual(const float* __restrict__ x, const float* __restrict__ res,
                              const float* __restrict__ g, const float* __restrict__ bb,
                              float* __restrict__ out, u16* __restrict__ outH, int M, int D) {
  int r = blockIdx.x * blockDim.x + threadIdx.x;
  if (r >= M) return;
  const float* xr = x + (size_t)r * D;
  const float* rr = res + (size_t)r * D;
  float s = 0.f, ss = 0.f;
  for (int d = 0; d < D; ++d) { float v = xr[d] + rr[d]; s += v; ss += v * v; }
  float mu = s / D, inv = rsqrtf(ss / D - mu * mu + 1e-5f);
  for (int d = 0; d < D; ++d) {
    float v = (xr[d] + rr[d] - mu) * inv * g[d] + bb[d];
    out[(size_t)r * D + d] = v;
    if (outH) outH[(size_t)r * D + d] = f2bf(v);
  }
}

// ---------------------------------------------------------------------------
// region-prior (softplus) * sigmoid gate
// ---------------------------------------------------------------------------
__global__ void k_gate(const float* __restrict__ tok, const float* __restrict__ g1w,
                       const float* __restrict__ g1b, const float* __restrict__ g2w,
                       const float* __restrict__ g2b, const float* __restrict__ rlogit,
                       float* __restrict__ outF, u16* __restrict__ outH) {
  int row = blockIdx.x * blockDim.x + threadIdx.x;
  if (row >= NTOK) return;
  int r = row % 6;
  const float* x = tok + (size_t)row * 64;
  float logit = g2b[0];
  for (int k = 0; k < 32; ++k) {
    float s = g1b[k];
    for (int c = 0; c < 64; ++c) s += x[c] * g1w[c * 32 + k];
    logit += fmaxf(s, 0.f) * g2w[k];
  }
  float gate = 1.f / (1.f + __expf(-logit));
  float rl = rlogit[r];
  float rw = (rl > 20.f) ? rl : log1pf(__expf(rl));      // softplus
  float scale = rw * gate;
  for (int d = 0; d < 64; ++d) {
    float v = x[d] * scale;
    outF[(size_t)row * 64 + d] = v;
    outH[(size_t)row * 64 + d] = f2bf(v);
  }
}

__global__ void k_qmean(const float* __restrict__ tokg, float* __restrict__ outF,
                        u16* __restrict__ outH) {
  int idx = blockIdx.x * blockDim.x + threadIdx.x;
  if (idx >= BT * 64) return;
  int fr = idx / 64, d = idx % 64;
  float s = 0.f;
  for (int j = 0; j < 6; ++j) s += tokg[(size_t)(fr * 6 + j) * 64 + d];
  s *= (1.f / 6.f);
  outF[idx] = s;
  outH[idx] = f2bf(s);
}

// ---------------------------------------------------------------------------
// gated global branch: tanh(alpha) * relu(LN(gf @ W + b))
// ---------------------------------------------------------------------------
__global__ void k_global_branch(const float* __restrict__ gf, const float* __restrict__ alpha,
                                const float* __restrict__ w, const float* __restrict__ bb,
                                const float* __restrict__ g, const float* __restrict__ be,
                                float* __restrict__ out) {
  int fr = blockIdx.x * blockDim.x + threadIdx.x;
  if (fr >= BT) return;
  float a0 = gf[fr * 4 + 0], a1 = gf[fr * 4 + 1], a2 = gf[fr * 4 + 2], a3 = gf[fr * 4 + 3];
  float y[64], s = 0.f, ss = 0.f;
  for (int d = 0; d < 64; ++d) {
    float v = a0 * w[d] + a1 * w[64 + d] + a2 * w[128 + d] + a3 * w[192 + d] + bb[d];
    y[d] = v; s += v; ss += v * v;
  }
  float mu = s / 64.f, inv = rsqrtf(ss / 64.f - mu * mu + 1e-5f);
  float ta = tanhf(alpha[0]);
  for (int d = 0; d < 64; ++d)
    out[(size_t)fr * 64 + d] = fmaxf((y[d] - mu) * inv * g[d] + be[d], 0.f) * ta;
}

// ---------------------------------------------------------------------------
// concat(fused_r, g) -> LN(128) -> + sinusoidal PE (computed on the fly)
// ---------------------------------------------------------------------------
__global__ void k_fuse(const float* __restrict__ frf, const float* __restrict__ gb,
                       const float* __restrict__ g, const float* __restrict__ bb,
                       float* __restrict__ outF, u16* __restrict__ outH) {
  int fr = blockIdx.x * blockDim.x + threadIdx.x;
  if (fr >= BT) return;
  float v[128], s = 0.f, ss = 0.f;
  for (int d = 0; d < 64; ++d) {
    v[d] = frf[(size_t)fr * 64 + d];
    v[64 + d] = gb[(size_t)fr * 64 + d];
  }
  for (int d = 0; d < 128; ++d) { s += v[d]; ss += v[d] * v[d]; }
  float mu = s / 128.f, inv = rsqrtf(ss / 128.f - mu * mu + 1e-5f);
  int t = fr % T_;
  const float c = -9.210340371976184f / 128.f;   // -ln(10000)/128
  for (int d = 0; d < 128; ++d) {
    float o = (v[d] - mu) * inv * g[d] + bb[d];
    float ang = (float)t * __expf((float)(2 * (d >> 1)) * c);
    o += (d & 1) ? __cosf(ang) : __sinf(ang);
    outF[(size_t)fr * 128 + d] = o;
    outH[(size_t)fr * 128 + d] = f2bf(o);
  }
}

// ---------------------------------------------------------------------------
// attention pooling over time (block per batch)
// ---------------------------------------------------------------------------
__launch_bounds__(256)
__global__ void k_attn_pool(const float* __restrict__ h, const float* __restrict__ wv,
                            const float* __restrict__ bv, float* __restrict__ pooled) {
  __shared__ float wt[256];
  __shared__ float stats[2];
  int b = blockIdx.x, t = threadIdx.x;
  const float* hr = h + (size_t)(b * T_ + t) * 128;
  float s = bv[0];
  for (int d = 0; d < 128; ++d) s += hr[d] * wv[d];
  wt[t] = s;
  __syncthreads();
  if (t == 0) {
    float mx = -3.0e38f;
    for (int j = 0; j < 256; ++j) mx = fmaxf(mx, wt[j]);
    float den = 0.f;
    for (int j = 0; j < 256; ++j) den += __expf(wt[j] - mx);
    stats[0] = mx; stats[1] = den;
  }
  __syncthreads();
  float w = __expf(wt[t] - stats[0]) / stats[1];
  __syncthreads();
  wt[t] = w;
  __syncthreads();
  if (t < 128) {
    float acc = 0.f;
    for (int j = 0; j < 256; ++j) acc += wt[j] * h[(size_t)(b * T_ + j) * 128 + t];
    pooled[b * 128 + t] = acc;
  }
}

// ---------------------------------------------------------------------------
// classifier head
// ---------------------------------------------------------------------------
__global__ void k_classifier(const float* __restrict__ pooled, const float* __restrict__ w1,
                             const float* __restrict__ b1, const float* __restrict__ lg,
                             const float* __restrict__ lb, const float* __restrict__ w2,
                             const float* __restrict__ b2, float* __restrict__ out) {
  int b = blockIdx.x * blockDim.x + threadIdx.x;
  if (b >= B_) return;
  const float* p = pooled + b * 128;
  float z[32], s = 0.f, ss = 0.f;
  for (int k = 0; k < 32; ++k) {
    float v = b1[k];
    for (int c = 0; c < 128; ++c) v += p[c] * w1[c * 32 + k];
    z[k] = v; s += v; ss += v * v;
  }
  float mu = s / 32.f, inv = rsqrtf(ss / 32.f - mu * mu + 1e-5f);
  for (int k = 0; k < 32; ++k) z[k] = fmaxf((z[k] - mu) * inv * lg[k] + lb[k], 0.f);
  for (int j = 0; j < 2; ++j) {
    float o = b2[j];
    for (int k = 0; k < 32; ++k) o += z[k] * w2[k * 2 + j];
    out[b * 2 + j] = o;
  }
}

// ---------------------------------------------------------------------------
// host launcher
// ---------------------------------------------------------------------------
extern "C" void kernel_launch(void* const* d_in, const int* in_sizes, int n_in,
                              void* d_out, int out_size, void* d_ws, size_t ws_size,
                              hipStream_t stream) {
  (void)in_sizes; (void)n_in; (void)out_size; (void)ws_size;
  const float* x0 = (const float*)d_in[0];
  const float* x1 = (const float*)d_in[1];
  const float* x2 = (const float*)d_in[2];
  const float* x3 = (const float*)d_in[3];
  const float* x4 = (const float*)d_in[4];
  const float* x5 = (const float*)d_in[5];
  const float* gfeat      = (const float*)d_in[6];
  const float* alpha      = (const float*)d_in[7];
  const float* gcn1_w     = (const float*)d_in[8];
  const float* gcn1_b     = (const float*)d_in[9];
  const float* gcn2_w     = (const float*)d_in[10];
  const float* gcn2_b     = (const float*)d_in[11];
  const float* rln_g      = (const float*)d_in[12];
  const float* rln_b      = (const float*)d_in[13];
  const float* sp_qkv_w   = (const float*)d_in[14];
  const float* sp_qkv_b   = (const float*)d_in[15];
  const float* sp_out_w   = (const float*)d_in[16];
  const float* sp_out_b   = (const float*)d_in[17];
  const float* sp_ln1_g   = (const float*)d_in[18];
  const float* sp_ln1_b   = (const float*)d_in[19];
  const float* sp_ff1_w   = (const float*)d_in[20];
  const float* sp_ff1_b   = (const float*)d_in[21];
  const float* sp_ff2_w   = (const float*)d_in[22];
  const float* sp_ff2_b   = (const float*)d_in[23];
  const float* sp_ln2_g   = (const float*)d_in[24];
  const float* sp_ln2_b   = (const float*)d_in[25];
  const float* region_log = (const float*)d_in[26];
  const float* gate1_w    = (const float*)d_in[27];
  const float* gate1_b    = (const float*)d_in[28];
  const float* gate2_w    = (const float*)d_in[29];
  const float* gate2_b    = (const float*)d_in[30];
  const float* ca_qkv_w   = (const float*)d_in[31];
  const float* ca_qkv_b   = (const float*)d_in[32];
  const float* ca_out_w   = (const float*)d_in[33];
  const float* ca_out_b   = (const float*)d_in[34];
  const float* ca_ln_g    = (const float*)d_in[35];
  const float* ca_ln_b    = (const float*)d_in[36];
  const float* glob_w     = (const float*)d_in[37];
  const float* glob_b     = (const float*)d_in[38];
  const float* glob_ln_g  = (const float*)d_in[39];
  const float* glob_ln_b  = (const float*)d_in[40];
  const float* fused_ln_g = (const float*)d_in[41];
  const float* fused_ln_b = (const float*)d_in[42];
  const float* tp_qkv_w   = (const float*)d_in[43];
  const float* tp_qkv_b   = (const float*)d_in[44];
  const float* tp_out_w   = (const float*)d_in[45];
  const float* tp_out_b   = (const float*)d_in[46];
  const float* tp_ln1_g   = (const float*)d_in[47];
  const float* tp_ln1_b   = (const float*)d_in[48];
  const float* tp_ff1_w   = (const float*)d_in[49];
  const float* tp_ff1_b   = (const float*)d_in[50];
  const float* tp_ff2_w   = (const float*)d_in[51];
  const float* tp_ff2_b   = (const float*)d_in[52];
  const float* tp_ln2_g   = (const float*)d_in[53];
  const float* tp_ln2_b   = (const float*)d_in[54];
  const float* attnproj_w = (const float*)d_in[55];
  const float* attnproj_b = (const float*)d_in[56];
  const float* cls1_w     = (const float*)d_in[57];
  const float* cls1_b     = (const float*)d_in[58];
  const float* cls_ln_g   = (const float*)d_in[59];
  const float* cls_ln_b   = (const float*)d_in[60];
  const float* cls2_w     = (const float*)d_in[61];
  const float* cls2_b     = (const float*)d_in[62];

  char* ws = (char*)d_ws;
  auto alloc = [&](size_t n) -> char* {
    char* p = ws;
    ws += (n + 255) & ~(size_t)255;
    return p;
  };

  float* tokF   = (float*)alloc((size_t)NTOK * 64 * 4);
  u16*   tokH   = (u16*)  alloc((size_t)NTOK * 64 * 2);
  float* sqkv   = (float*)alloc((size_t)NTOK * 192 * 4);   // reused as CA kv
  u16*   attnH  = (u16*)  alloc((size_t)NTOK * 64 * 2);
  float* projF  = (float*)alloc((size_t)NTOK * 64 * 4);
  float* x1F    = (float*)alloc((size_t)NTOK * 64 * 4);
  u16*   x1H    = (u16*)  alloc((size_t)NTOK * 64 * 2);
  float* tok2F  = (float*)alloc((size_t)NTOK * 64 * 4);
  float* tokgF  = (float*)alloc((size_t)NTOK * 64 * 4);
  u16*   tokgH  = (u16*)  alloc((size_t)NTOK * 64 * 2);
  float* qmF    = (float*)alloc((size_t)BT * 64 * 4);
  u16*   qmH    = (u16*)  alloc((size_t)BT * 64 * 2);
  float* qqF    = (float*)alloc((size_t)BT * 192 * 4);
  u16*   cattnH = (u16*)  alloc((size_t)BT * 64 * 2);
  float* cprojF = (float*)alloc((size_t)BT * 64 * 4);
  float* frF    = (float*)alloc((size_t)BT * 64 * 4);
  float* gbF    = (float*)alloc((size_t)BT * 64 * 4);
  float* fusedF = (float*)alloc((size_t)BT * 128 * 4);
  u16*   fusedH = (u16*)  alloc((size_t)BT * 128 * 2);
  float* tqkv   = (float*)alloc((size_t)BT * 384 * 4);
  u16*   tattnH = (u16*)  alloc((size_t)BT * 128 * 2);
  float* tprojF = (float*)alloc((size_t)BT * 128 * 4);
  float* h1F    = (float*)alloc((size_t)BT * 128 * 4);
  u16*   h1H    = (u16*)  alloc((size_t)BT * 128 * 2);
  float* hfF    = (float*)alloc((size_t)BT * 128 * 4);
  float* pooled = (float*)alloc((size_t)B_ * 128 * 4);

  u16* sp_qkv_wH = (u16*)alloc(64 * 192 * 2);
  u16* sp_out_wH = (u16*)alloc(64 * 64 * 2);
  u16* sp_f1H    = (u16*)alloc(64 * 2048 * 2);
  u16* sp_f2H    = (u16*)alloc(2048 * 64 * 2);
  u16* ca_qkv_wH = (u16*)alloc(64 * 192 * 2);
  u16* ca_out_wH = (u16*)alloc(64 * 64 * 2);
  u16* tp_qkv_wH = (u16*)alloc(128 * 384 * 2);
  u16* tp_out_wH = (u16*)alloc(128 * 128 * 2);
  u16* tp_f1H    = (u16*)alloc(128 * 2048 * 2);
  u16* tp_f2H    = (u16*)alloc(2048 * 128 * 2);

  auto cvt = [&](const float* src, u16* dst, int n) {
    k_f32_to_bf16<<<(n + 255) / 256, 256, 0, stream>>>(src, dst, n);
  };
  cvt(sp_qkv_w, sp_qkv_wH, 64 * 192);
  cvt(sp_out_w, sp_out_wH, 64 * 64);
  cvt(sp_ff1_w, sp_f1H, 64 * 2048);
  cvt(sp_ff2_w, sp_f2H, 2048 * 64);
  cvt(ca_qkv_w, ca_qkv_wH, 64 * 192);
  cvt(ca_out_w, ca_out_wH, 64 * 64);
  cvt(tp_qkv_w, tp_qkv_wH, 128 * 384);
  cvt(tp_out_w, tp_out_wH, 128 * 128);
  cvt(tp_ff1_w, tp_f1H, 128 * 2048);
  cvt(tp_ff2_w, tp_f2H, 2048 * 128);

  // 1. per-region GCN -> tokens (NTOK x 64), LN'd
  k_gcn<<<NTOK / 8, 256, 0, stream>>>(x0, x1, x2, x3, x4, x5, gcn1_w, gcn1_b, gcn2_w, gcn2_b,
                                      rln_g, rln_b, tokF, tokH);
  // 2. spatial QKV projection
  k_gemm_bf16<<<dim3(192 / 64, NTOK / 64), 256, 0, stream>>>(tokH, sp_qkv_wH, sp_qkv_b, sqkv,
                                                             NTOK, 192, 64, 0);
  // 3. spatial attention (6 tokens/frame)
  k_spatial_attn<<<(BT * NH * 6 + 255) / 256, 256, 0, stream>>>(sqkv, attnH);
  // 4. out projection
  k_gemm_bf16<<<dim3(1, NTOK / 64), 256, 0, stream>>>(attnH, sp_out_wH, sp_out_b, projF,
                                                      NTOK, 64, 64, 0);
  // 5. x1 = LN(tok + proj)
  k_ln_residual<<<(NTOK + 255) / 256, 256, 0, stream>>>(projF, tokF, sp_ln1_g, sp_ln1_b,
                                                        x1F, x1H, NTOK, 64);
  // 6. fused FFN (64 -> 2048 -> 64) + residual + LN
  k_ffn_fused<64><<<NTOK / 64, 256, 0, stream>>>(x1H, x1F, sp_f1H, sp_ff1_b, sp_f2H, sp_ff2_b,
                                                 sp_ln2_g, sp_ln2_b, tok2F);
  // 7. region priors * gates
  k_gate<<<(NTOK + 255) / 256, 256, 0, stream>>>(tok2F, gate1_w, gate1_b, gate2_w, gate2_b,
                                                 region_log, tokgF, tokgH);
  // 8. per-frame mean query
  k_qmean<<<(BT * 64 + 255) / 256, 256, 0, stream>>>(tokgF, qmF, qmH);
  // 9. CA K/V over tokens (reuse sqkv)
  k_gemm_bf16<<<dim3(3, NTOK / 64), 256, 0, stream>>>(tokgH, ca_qkv_wH, ca_qkv_b, sqkv,
                                                      NTOK, 192, 64, 0);
  // 10. CA Q over mean queries
  k_gemm_bf16<<<dim3(3, BT / 64), 256, 0, stream>>>(qmH, ca_qkv_wH, ca_qkv_b, qqF,
                                                    BT, 192, 64, 0);
  // 11. cross-attention
  k_ca_attn<<<(BT * NH + 255) / 256, 256, 0, stream>>>(qqF, sqkv, cattnH);
  // 12. CA out projection
  k_gemm_bf16<<<dim3(1, BT / 64), 256, 0, stream>>>(cattnH, ca_out_wH, ca_out_b, cprojF,
                                                    BT, 64, 64, 0);
  // 13. fused_r = LN(qmean + proj)
  k_ln_residual<<<(BT + 255) / 256, 256, 0, stream>>>(cprojF, qmF, ca_ln_g, ca_ln_b,
                                                      frF, nullptr, BT, 64);
  // 14. gated global branch
  k_global_branch<<<(BT + 255) / 256, 256, 0, stream>>>(gfeat, alpha, glob_w, glob_b,
                                                        glob_ln_g, glob_ln_b, gbF);
  // 15. concat + LN + positional encoding
  k_fuse<<<(BT + 255) / 256, 256, 0, stream>>>(frF, gbF, fused_ln_g, fused_ln_b, fusedF, fusedH);
  // 16. temporal QKV projection
  k_gemm_bf16<<<dim3(384 / 64, BT / 64), 256, 0, stream>>>(fusedH, tp_qkv_wH, tp_qkv_b, tqkv,
                                                           BT, 384, 128, 0);
  // 17. temporal attention (T=256, online softmax)
  k_temporal_attn<<<B_ * NH, 256, 0, stream>>>(tqkv, tattnH);
  // 18. temporal out projection
  k_gemm_bf16<<<dim3(2, BT / 64), 256, 0, stream>>>(tattnH, tp_out_wH, tp_out_b, tprojF,
                                                    BT, 128, 128, 0);
  // 19. h1 = LN(fused + proj)
  k_ln_residual<<<(BT + 255) / 256, 256, 0, stream>>>(tprojF, fusedF, tp_ln1_g, tp_ln1_b,
                                                      h1F, h1H, BT, 128);
  // 20. fused FFN (128 -> 2048 -> 128) + residual + LN
  k_ffn_fused<128><<<BT / 64, 256, 0, stream>>>(h1H, h1F, tp_f1H, tp_ff1_b, tp_f2H, tp_ff2_b,
                                                tp_ln2_g, tp_ln2_b, hfF);
  // 21. attention pooling over time
  k_attn_pool<<<B_, 256, 0, stream>>>(hfF, attnproj_w, attnproj_b, pooled);
  // 22. classifier head
  k_classifier<<<1, 32, 0, stream>>>(pooled, cls1_w, cls1_b, cls_ln_g, cls_ln_b,
                                     cls2_w, cls2_b, (float*)d_out);
}
